// OptimizedMoE_29901562315094
// MI455X (gfx1250) — matrix operand order repaired and checked by reading, not compiled, not run on hardware
//
#include <hip/hip_runtime.h>

// Problem constants (from reference): B=8, T=2048 -> NT tokens
#define NT      16384
#define C_DIM   896
#define RH_DIM  448
#define E_NUM   3
#define H_DIM   3136
#define TM      32            // tokens per expert block (2 x 16-row WMMA tiles)

typedef __attribute__((ext_vector_type(16))) __bf16       v16bf;
typedef __attribute__((ext_vector_type(8)))  float        v8f;
typedef __attribute__((ext_vector_type(4)))  unsigned int uint4v;

union V16B { v16bf bf; uint4v u[2]; };

// D = A(16x32 bf16) * B(32x16 bf16) + C(16x16 f32)
__device__ __forceinline__ v8f wmma_bf16(const V16B& a, const V16B& b, v8f c) {
    return __builtin_amdgcn_wmma_f32_16x16x32_bf16(
        /*neg_a=*/false, a.bf, /*neg_b=*/false, b.bf,
        /*c_mod=*/(short)0, c, /*reuse_a=*/false, /*reuse_b=*/false);
}

// 16 bf16 elements: [0..7] from p0, [8..15] from p1 (each 16B-aligned)
__device__ __forceinline__ V16B load2x8(const __bf16* p0, const __bf16* p1) {
    V16B r;
    r.u[0] = *(const uint4v*)p0;
    r.u[1] = *(const uint4v*)p1;
    return r;
}

// ---------------- counters ----------------
__global__ void k_zero(int* counts) {
    if (threadIdx.x < E_NUM) counts[threadIdx.x] = 0;
}

// ---------------- fp32 -> bf16 transposed weight conversion ----------------
// rw1 [C,RH] -> rw1t [RH,C]
__global__ void k_tr_rw1(const float* __restrict__ rw1, __bf16* __restrict__ out) {
    int i = blockIdx.x * 256 + threadIdx.x;
    if (i >= RH_DIM * C_DIM) return;
    int n = i / C_DIM, k = i - n * C_DIM;
    out[i] = (__bf16)rw1[k * RH_DIM + n];
}
// ew_in [E,C,H] -> wint [E,H,C]
__global__ void k_tr_win(const float* __restrict__ win, __bf16* __restrict__ out) {
    long i = (long)blockIdx.x * 256 + threadIdx.x;
    if (i >= (long)E_NUM * H_DIM * C_DIM) return;
    long e = i / ((long)H_DIM * C_DIM);
    long r = i - e * (long)H_DIM * C_DIM;
    long n = r / C_DIM, k = r - n * C_DIM;
    out[i] = (__bf16)win[(e * C_DIM + k) * H_DIM + n];
}
// ew_out [E,H,C] -> woutt [E,C,H]
__global__ void k_tr_wout(const float* __restrict__ wout, __bf16* __restrict__ out) {
    long i = (long)blockIdx.x * 256 + threadIdx.x;
    if (i >= (long)E_NUM * C_DIM * H_DIM) return;
    long e = i / ((long)C_DIM * H_DIM);
    long r = i - e * (long)C_DIM * H_DIM;
    long n = r / H_DIM, k = r - n * H_DIM;
    out[i] = (__bf16)wout[(e * H_DIM + k) * C_DIM + n];
}

// ---------------- router: WMMA hidden GEMM + argmax + compaction ----------------
// block = 128 threads (4 waves), 16 tokens per block, 2-way N register blocking.
__global__ void k_router(const float* __restrict__ x, const __bf16* __restrict__ rw1t,
                         const float* __restrict__ rb1, const float* __restrict__ rw2,
                         const float* __restrict__ rb2, int* counts, int* lists) {
    extern __shared__ char smem[];
    __bf16* s_x   = (__bf16*)smem;                                       // 16 x 896 bf16
    float*  s_hid = (float*)(smem + 16 * C_DIM * 2);                     // 16 x 448 f32
    float*  s_log = (float*)(smem + 16 * C_DIM * 2 + 16 * RH_DIM * 4);   // 16 x 3 f32
    const int t0  = blockIdx.x * 16;
    const int tid = threadIdx.x;

    for (int i = tid; i < 16 * C_DIM; i += 128) {
        int r = i / C_DIM, c = i - r * C_DIM;
        s_x[i] = (__bf16)x[(size_t)(t0 + r) * C_DIM + c];
    }
    __syncthreads();

    const int w    = tid >> 5, lane = tid & 31;
    const int nlo  = lane & 15;
    const int aoff = (lane < 16) ? 0 : 8;    // A K-chunk select (ISA 7.12.2)
    const int boff = (lane < 16) ? 0 : 8;    // B K-chunk select (same as A)

    for (int nt0 = w * 2; nt0 < RH_DIM / 16; nt0 += 8) {   // pairs cover 28 tiles exactly
        v8f acc[2];
        const __bf16* brow[2];
        #pragma unroll
        for (int q = 0; q < 2; ++q) {
            int n = (nt0 + q) * 16 + nlo;
            float bias = rb1[n];
            #pragma unroll
            for (int i = 0; i < 8; ++i) acc[q][i] = bias;
            brow[q] = rw1t + (size_t)n * C_DIM;
        }
        for (int kb = 0; kb < C_DIM; kb += 32) {
            // issue all fragment loads first, then compute
            V16B a = load2x8(s_x + nlo * C_DIM + kb + aoff,
                             s_x + nlo * C_DIM + kb + 16 + aoff);
            V16B b[2];
            #pragma unroll
            for (int q = 0; q < 2; ++q)
                b[q] = load2x8(brow[q] + kb + boff, brow[q] + kb + 16 + boff);
            #pragma unroll
            for (int q = 0; q < 2; ++q)
                acc[q] = wmma_bf16(a, b[q], acc[q]);
        }
        #pragma unroll
        for (int q = 0; q < 2; ++q) {
            int n = (nt0 + q) * 16 + nlo;
            #pragma unroll
            for (int r = 0; r < 8; ++r) {
                int m = (lane < 16) ? r : (8 + r);
                float v = acc[q][r];
                s_hid[m * RH_DIM + n] = v > 0.f ? v : 0.f;
            }
        }
    }
    __syncthreads();

    if (tid < 16 * E_NUM) {   // 48 threads: one (token, expert) dot each
        int m = tid & 15, e = tid >> 4;
        float s = rb2[e];
        for (int k = 0; k < RH_DIM; ++k) s += s_hid[m * RH_DIM + k] * rw2[k * E_NUM + e];
        s_log[m * E_NUM + e] = s;
    }
    __syncthreads();

    if (tid < 16) {   // argmax (ties -> lowest index, matching top_k) + compaction
        float l0 = s_log[tid * 3 + 0], l1 = s_log[tid * 3 + 1], l2 = s_log[tid * 3 + 2];
        int best = 0; float bv = l0;
        if (l1 > bv) { bv = l1; best = 1; }
        if (l2 > bv) { bv = l2; best = 2; }
        int pos = atomicAdd(&counts[best], 1);
        lists[(size_t)best * NT + pos] = t0 + tid;
    }
}

// ---------------- fused expert MLP: h=ReLU(xW_in+b); out = x + hW_out + b ----------------
// grid = (NT/TM, E), block = 256 (8 waves). 32 tokens per block = two 16-row A tiles,
// 2(M) x 4(N) register blocking -> 8 WMMAs per fragment load set (issued as one clause).
// Hidden tile (32 x 3136 bf16, 196 KB) lives in LDS (CDNA5: 320 KB/WGP); its space
// doubles as the fp32 async-staging buffer for the gathered x rows before GEMM1 runs.
__global__ void k_expert(const float* __restrict__ x, const __bf16* __restrict__ wint,
                         const float* __restrict__ bin, const __bf16* __restrict__ woutt,
                         const float* __restrict__ bout, const int* __restrict__ counts,
                         const int* __restrict__ lists, float* __restrict__ out) {
    const int e   = blockIdx.y;
    const int cnt = counts[e];
    const int m0  = blockIdx.x * TM;
    if (m0 >= cnt) return;

    extern __shared__ char smem[];
    __bf16* s_x   = (__bf16*)smem;                                   // 32 x 896 bf16  (56 KB)
    __bf16* s_h   = (__bf16*)(smem + TM * C_DIM * 2);                // 32 x 3136 bf16 (196 KB)
    int*    s_tok = (int*)(smem + TM * C_DIM * 2 + TM * H_DIM * 2);  // 32 ints
    float*  s_xf  = (float*)s_h;                                     // fp32 staging overlay (112 KB)
    const int tid = threadIdx.x;

    if (tid < TM) {
        int idx = m0 + tid;
        s_tok[tid] = lists[(size_t)e * NT + ((idx < cnt) ? idx : m0)];
    }
    __syncthreads();

    // Gather x rows global -> LDS via CDNA5 async-to-LDS DMA (no VGPR landing zone),
    // then convert fp32 -> bf16 in LDS.
    {
        const int CHUNKS = TM * C_DIM / 4;       // 7168 x 16B
        for (int i = tid; i < CHUNKS; i += 256) {
            int r  = i / (C_DIM / 4);
            int c4 = i - r * (C_DIM / 4);
            const float* gp = x + (size_t)s_tok[r] * C_DIM + (size_t)c4 * 4;
            unsigned lds_off = (unsigned)(unsigned long long)(void*)(s_xf + (size_t)i * 4);
            asm volatile("global_load_async_to_lds_b128 %0, %1, off"
                         :: "v"(lds_off), "v"(gp) : "memory");
        }
        asm volatile("s_wait_asynccnt 0x0" ::: "memory");
        __syncthreads();
        for (int i = tid; i < TM * C_DIM; i += 256) s_x[i] = (__bf16)s_xf[i];
        __syncthreads();
    }

    const int w    = tid >> 5, lane = tid & 31;
    const int nlo  = lane & 15;
    const int aoff = (lane < 16) ? 0 : 8;
    const int boff = (lane < 16) ? 0 : 8;

    // GEMM1: [32 x 896] x [896 x 3136] -> s_h (ReLU, bf16). 196 n-tiles = 49 quads / 8 waves.
    for (int nt0 = w * 4; nt0 < H_DIM / 16; nt0 += 32) {
        v8f acc[2][4];
        const __bf16* brow[4];
        #pragma unroll
        for (int q = 0; q < 4; ++q) {
            int n = (nt0 + q) * 16 + nlo;
            float bias = bin[(size_t)e * H_DIM + n];
            #pragma unroll
            for (int i = 0; i < 8; ++i) { acc[0][q][i] = bias; acc[1][q][i] = bias; }
            brow[q] = wint + ((size_t)e * H_DIM + n) * C_DIM;
        }
        for (int kb = 0; kb < C_DIM; kb += 32) {
            __builtin_prefetch(brow[0] + kb + 128, 0, 0);   // global_prefetch_b8
            // issue all fragment loads first (one clause), then 8 back-to-back WMMAs
            V16B a0 = load2x8(s_x + nlo * C_DIM + kb + aoff,
                              s_x + nlo * C_DIM + kb + 16 + aoff);
            V16B a1 = load2x8(s_x + (16 + nlo) * C_DIM + kb + aoff,
                              s_x + (16 + nlo) * C_DIM + kb + 16 + aoff);
            V16B b[4];
            #pragma unroll
            for (int q = 0; q < 4; ++q)
                b[q] = load2x8(brow[q] + kb + boff, brow[q] + kb + 16 + boff);
            #pragma unroll
            for (int q = 0; q < 4; ++q) {
                acc[0][q] = wmma_bf16(a0, b[q], acc[0][q]);
                acc[1][q] = wmma_bf16(a1, b[q], acc[1][q]);
            }
        }
        #pragma unroll
        for (int h = 0; h < 2; ++h) {
            #pragma unroll
            for (int q = 0; q < 4; ++q) {
                int n = (nt0 + q) * 16 + nlo;
                #pragma unroll
                for (int r = 0; r < 8; ++r) {
                    int m = h * 16 + ((lane < 16) ? r : (8 + r));
                    float v = acc[h][q][r];
                    s_h[m * H_DIM + n] = (__bf16)(v > 0.f ? v : 0.f);
                }
            }
        }
    }
    __syncthreads();

    // GEMM2: [32 x 3136] x [3136 x 896] + residual -> out. 56 n-tiles = 14 quads / 8 waves.
    for (int ct0 = w * 4; ct0 < C_DIM / 16; ct0 += 32) {
        v8f acc[2][4];
        const __bf16* brow[4];
        #pragma unroll
        for (int q = 0; q < 4; ++q) {
            int n = (ct0 + q) * 16 + nlo;
            float bias = bout[(size_t)e * C_DIM + n];
            #pragma unroll
            for (int i = 0; i < 8; ++i) { acc[0][q][i] = bias; acc[1][q][i] = bias; }
            brow[q] = woutt + ((size_t)e * C_DIM + n) * H_DIM;
        }
        for (int kb = 0; kb < H_DIM; kb += 32) {
            __builtin_prefetch(brow[0] + kb + 128, 0, 0);
            V16B a0 = load2x8(s_h + nlo * H_DIM + kb + aoff,
                              s_h + nlo * H_DIM + kb + 16 + aoff);
            V16B a1 = load2x8(s_h + (16 + nlo) * H_DIM + kb + aoff,
                              s_h + (16 + nlo) * H_DIM + kb + 16 + aoff);
            V16B b[4];
            #pragma unroll
            for (int q = 0; q < 4; ++q)
                b[q] = load2x8(brow[q] + kb + boff, brow[q] + kb + 16 + boff);
            #pragma unroll
            for (int q = 0; q < 4; ++q) {
                acc[0][q] = wmma_bf16(a0, b[q], acc[0][q]);
                acc[1][q] = wmma_bf16(a1, b[q], acc[1][q]);
            }
        }
        #pragma unroll
        for (int h = 0; h < 2; ++h) {
            #pragma unroll
            for (int q = 0; q < 4; ++q) {
                int n = (ct0 + q) * 16 + nlo;
                #pragma unroll
                for (int r = 0; r < 8; ++r) {
                    int m = h * 16 + ((lane < 16) ? r : (8 + r));
                    if (m0 + m < cnt) {
                        size_t tok = (size_t)s_tok[m];
                        // single-use output: non-temporal store (don't dirty near caches)
                        __builtin_nontemporal_store(x[tok * C_DIM + n] + acc[h][q][r],
                                                    &out[tok * C_DIM + n]);
                    }
                }
            }
        }
    }
}

extern "C" void kernel_launch(void* const* d_in, const int* in_sizes, int n_in,
                              void* d_out, int out_size, void* d_ws, size_t ws_size,
                              hipStream_t stream) {
    const float* x      = (const float*)d_in[0];
    const float* rw1    = (const float*)d_in[1];
    const float* rb1    = (const float*)d_in[2];
    const float* rw2    = (const float*)d_in[3];
    const float* rb2    = (const float*)d_in[4];
    const float* ew_in  = (const float*)d_in[5];
    const float* eb_in  = (const float*)d_in[6];
    const float* ew_out = (const float*)d_in[7];
    const float* eb_out = (const float*)d_in[8];
    float* out = (float*)d_out;

    char* ws = (char*)d_ws;
    size_t off = 0;
    __bf16* rw1t  = (__bf16*)(ws + off); off += (size_t)RH_DIM * C_DIM * 2;          // 0.8 MB
    __bf16* wint  = (__bf16*)(ws + off); off += (size_t)E_NUM * H_DIM * C_DIM * 2;   // 16.9 MB
    __bf16* woutt = (__bf16*)(ws + off); off += (size_t)E_NUM * C_DIM * H_DIM * 2;   // 16.9 MB
    int* counts   = (int*)(ws + off);    off += 16;
    int* lists    = (int*)(ws + off);    off += (size_t)E_NUM * NT * 4;              // 0.2 MB

    k_zero<<<1, 32, 0, stream>>>(counts);
    k_tr_rw1<<<(RH_DIM * C_DIM + 255) / 256, 256, 0, stream>>>(rw1, rw1t);
    {
        long n = (long)E_NUM * H_DIM * C_DIM;
        k_tr_win<<<(unsigned)((n + 255) / 256), 256, 0, stream>>>(ew_in, wint);
        k_tr_wout<<<(unsigned)((n + 255) / 256), 256, 0, stream>>>(ew_out, woutt);
    }
    size_t router_smem = (size_t)16 * C_DIM * 2 + (size_t)16 * RH_DIM * 4 + 16 * E_NUM * 4;
    k_router<<<NT / 16, 128, router_smem, stream>>>(x, rw1t, rb1, rw2, rb2, counts, lists);

    size_t expert_smem = (size_t)TM * C_DIM * 2 + (size_t)TM * H_DIM * 2 + TM * 4; // ~252 KB (CDNA5: 320 KB/WGP)
    k_expert<<<dim3(NT / TM, E_NUM), 256, expert_smem, stream>>>(
        x, wint, eb_in, woutt, eb_out, counts, lists, out);
}
